// CrossAttention_62148176773697
// MI455X (gfx1250) — matrix operand order repaired and checked by reading, not compiled, not run on hardware
//
#include <hip/hip_runtime.h>

// CrossAttention (local neighborhood attention) for MI455X / gfx1250, wave32 WMMA.
// B=4, CIN=64, N=8192, K=32, COUT=64, H=8, D=8.

typedef __attribute__((ext_vector_type(16))) _Float16 v16h;
typedef __attribute__((ext_vector_type(8)))  float    v8f;
typedef __attribute__((ext_vector_type(4)))  float    v4f;

#define NPTS   8192
#define KNBR   32
#define CHAN   64

// ds_swizzle xor-mode: offset = {1'b0, xor[14:10], or[9:5], and[4:0]}
#define SWZ_XOR(v, m) __builtin_amdgcn_ds_swizzle((v), (0x1F | ((m) << 10)))

__device__ __forceinline__ float xsum16(float x) {
  x += __int_as_float(SWZ_XOR(__float_as_int(x), 1));
  x += __int_as_float(SWZ_XOR(__float_as_int(x), 2));
  x += __int_as_float(SWZ_XOR(__float_as_int(x), 4));
  x += __int_as_float(SWZ_XOR(__float_as_int(x), 8));
  return x;
}
__device__ __forceinline__ float xmax16(float x) {
  x = fmaxf(x, __int_as_float(SWZ_XOR(__float_as_int(x), 1)));
  x = fmaxf(x, __int_as_float(SWZ_XOR(__float_as_int(x), 2)));
  x = fmaxf(x, __int_as_float(SWZ_XOR(__float_as_int(x), 4)));
  x = fmaxf(x, __int_as_float(SWZ_XOR(__float_as_int(x), 8)));
  return x;
}

__device__ __forceinline__ v8f wmma_f16(v16h a, v16h b, v8f c) {
  return __builtin_amdgcn_wmma_f32_16x16x32_f16(false, a, false, b, (short)0, c,
                                                false, false);
}

// ---------------------------------------------------------------------------
// Kernel 1: q = (1/sqrt(D)) * q_w @ pcd  -> qws[point][channel]  (fp32, 8 MB)
// One wave per 16 points. 8 waves / block.
// ---------------------------------------------------------------------------
__global__ void __launch_bounds__(256)
qproj_kernel(const float* __restrict__ pcd,
             const float* __restrict__ q_w,
             float* __restrict__ qws) {
  __shared__ __align__(16) float qlds[8][1088];  // 16 pts * 68-stride (pad vs bank conflicts)

  const int lane = threadIdx.x & 31;
  const int wv   = threadIdx.x >> 5;
  const int w    = blockIdx.x * 8 + wv;
  const int gp0  = w * 16;            // first global point of this wave
  const int b    = gp0 >> 13;         // /8192
  const int n0   = gp0 & (NPTS - 1);
  const int dl   = lane >> 4;         // 0 or 1 (lane half)
  const int j    = lane & 15;

  // B tiles: pcd columns (rows = input channel c, cols = 16 points).
  // wave32 16-bit B layout (matches A): elem e -> K = 16*(e/8) + 8*(lane/16)
  // + (e&7); column = lane&15.
  v16h Bt[2];
  #pragma unroll
  for (int s = 0; s < 2; ++s) {
    v16h bb;
    #pragma unroll
    for (int e = 0; e < 16; ++e) {
      const int c = 32 * s + 16 * (e >> 3) + 8 * dl + (e & 7);
      bb[e] = (_Float16)pcd[(((long)(b * CHAN + c)) << 13) + n0 + j];
    }
    Bt[s] = bb;
  }

  v8f C[4] = {};
  #pragma unroll
  for (int m = 0; m < 4; ++m) {
    #pragma unroll
    for (int s = 0; s < 2; ++s) {
      // A tile: 16-bit A layout (ISA 7.12.2): elem e -> K = 16*(e/8) + 8*(lane/16)
      // + 2*((e/2)&3) + (e&1); row M = lane&15.
      v16h A;
      #pragma unroll
      for (int e = 0; e < 16; ++e) {
        const int c = 32 * s + 16 * (e >> 3) + 8 * dl + 2 * ((e >> 1) & 3) + (e & 1);
        const int o = 16 * m + j;
        A[e] = (_Float16)q_w[o * CHAN + c];
      }
      C[m] = wmma_f16(A, Bt[s], C[m]);
    }
  }

  // C layout: VGPR r -> channel o = 16m + 8*dl + r, column = point (lane&15).
  const float invs = 0.35355339059327373f;  // 1/sqrt(8)
  #pragma unroll
  for (int m = 0; m < 4; ++m)
    #pragma unroll
    for (int r = 0; r < 8; ++r)
      qlds[wv][j * 68 + 16 * m + 8 * dl + r] = C[m][r] * invs;
  __syncthreads();

  // Coalesced store: wave writes its contiguous 1024 floats of qws.
  float* dst = qws + (long)gp0 * CHAN;
  #pragma unroll
  for (int i = 0; i < 8; ++i) {
    const int g = i * 128 + lane * 4;           // 0..1023, step 4 floats
    const int p = g >> 6;                       // point-in-tile
    const v4f val = *(const v4f*)&qlds[wv][p * 68 + j * 4];
    *(v4f*)(dst + g) = val;
  }
}

// ---------------------------------------------------------------------------
// Kernel 2: per point -- k/v projections via WMMA (neighbors read ONCE as
// shared B-tiles), softmax over K=32 neighbors, aggregate, coalesced store.
// One wave per point, 8 waves/block (8 consecutive points, same batch).
// ---------------------------------------------------------------------------
__global__ void __launch_bounds__(256)
attn_kernel(const float* __restrict__ neighbors,
            const float* __restrict__ k_w,
            const float* __restrict__ v_w,
            const float* __restrict__ qws,
            float* __restrict__ out) {
  // Weights pre-swizzled into the per-lane A-tile layout: [proj][m][s][lane][e]
  __shared__ __align__(32) _Float16 wlds[2 * 4 * 2 * 32 * 16];  // 16 KB
  __shared__ __align__(16) float xlds[8][64];                   // 2 KB output staging

  const int t = threadIdx.x;

  // Stage k_w / v_w into LDS once per block (served from L2, one-time cost).
  #pragma unroll
  for (int i = 0; i < 32; ++i) {
    const int f  = t * 32 + i;
    const int e  = f & 15;
    const int ln = (f >> 4) & 31;
    const int s  = (f >> 9) & 1;
    const int m  = (f >> 10) & 3;
    const int pj = f >> 12;
    const int c  = 32 * s + 16 * (e >> 3) + 8 * (ln >> 4) + 2 * ((e >> 1) & 3) + (e & 1);
    const int o  = 16 * m + (ln & 15);
    const float* W = pj ? v_w : k_w;
    wlds[f] = (_Float16)W[o * CHAN + c];
  }
  __syncthreads();

  const int lane = t & 31;
  const int wv   = t >> 5;
  const int gp   = blockIdx.x * 8 + wv;   // global point index
  const int b    = gp >> 13;
  const int n    = gp & (NPTS - 1);
  const int dl   = lane >> 4;
  const int j    = lane & 15;

  // ---- Load neighbor slab as four 32x16 B-tiles (fp32 -> fp16), read ONCE.
  const float* nb = neighbors + (long)b * CHAN * (long)NPTS * KNBR + (long)n * KNBR;
  v16h Bt[4];  // [s*2 + ntile]
  #pragma unroll
  for (int s = 0; s < 2; ++s) {
    #pragma unroll
    for (int tt = 0; tt < 2; ++tt) {
      v16h bb;
      #pragma unroll
      for (int e = 0; e < 16; ++e) {
        const int c = 32 * s + 16 * (e >> 3) + 8 * dl + (e & 7);  // B row (input channel)
        bb[e] = (_Float16)nb[(long)c * ((long)NPTS * KNBR) + 16 * tt + j];
      }
      Bt[s * 2 + tt] = bb;
    }
  }

  // ---- K projection: Ck[o, k'] for o=0..63, k'=0..31  (16 WMMAs)
  v8f Ck[8] = {};
  #pragma unroll
  for (int m = 0; m < 4; ++m) {
    #pragma unroll
    for (int s = 0; s < 2; ++s) {
      const v16h A = *(const v16h*)&wlds[((0 * 4 + m) * 2 + s) * 512 + lane * 16];
      #pragma unroll
      for (int tt = 0; tt < 2; ++tt)
        Ck[m * 2 + tt] = wmma_f16(A, Bt[s * 2 + tt], Ck[m * 2 + tt]);
    }
  }

  // ---- Energies + softmax. Lane holds channels o = 16m + 8*dl + r, i.e. a
  // full head h = 2m + dl per m-tile, for neighbor column k' = 16*tt + j.
  const float* qp = qws + (long)gp * CHAN;  // q pre-scaled by 1/sqrt(D)
  float w0[4], w1[4];
  #pragma unroll
  for (int m = 0; m < 4; ++m) {
    const v4f qlo = *(const v4f*)(qp + 16 * m + 8 * dl);
    const v4f qhi = *(const v4f*)(qp + 16 * m + 8 * dl + 4);
    float e0 = 0.f, e1 = 0.f;
    #pragma unroll
    for (int r = 0; r < 4; ++r) {
      e0 += qlo[r] * Ck[2 * m][r]     + qhi[r] * Ck[2 * m][r + 4];
      e1 += qlo[r] * Ck[2 * m + 1][r] + qhi[r] * Ck[2 * m + 1][r + 4];
    }
    const float mx = xmax16(fmaxf(e0, e1));         // max over 32 neighbors
    const float a0 = __expf(e0 - mx);
    const float a1 = __expf(e1 - mx);
    const float inv = 1.0f / xsum16(a0 + a1);       // sum over 32 neighbors
    w0[m] = a0 * inv;
    w1[m] = a1 * inv;
  }

  // ---- V projection + attention-weighted aggregation (B-tiles reused).
  #pragma unroll
  for (int m = 0; m < 4; ++m) {
    v8f Cv0 = {}, Cv1 = {};
    #pragma unroll
    for (int s = 0; s < 2; ++s) {
      const v16h A = *(const v16h*)&wlds[((1 * 4 + m) * 2 + s) * 512 + lane * 16];
      Cv0 = wmma_f16(A, Bt[s * 2 + 0], Cv0);
      Cv1 = wmma_f16(A, Bt[s * 2 + 1], Cv1);
    }
    #pragma unroll
    for (int r = 0; r < 8; ++r) {
      float p = w0[m] * Cv0[r] + w1[m] * Cv1[r];  // partial over this lane's 2 cols
      p = xsum16(p);                              // sum over all 32 neighbors
      if (j == 0) xlds[wv][16 * m + 8 * dl + r] = p;
    }
  }
  __syncthreads();

  // ---- Coalesced store: 8 points x 64 channels, out[b][o][n].
  const int b0 = (blockIdx.x * 8) >> 13;
  const int nb0 = (blockIdx.x * 8) & (NPTS - 1);
  #pragma unroll
  for (int i = 0; i < 2; ++i) {
    const int idx = t + 256 * i;        // 0..511
    const int o   = idx >> 3;
    const int pl  = idx & 7;
    out[((long)b0 * CHAN + o) * NPTS + nb0 + pl] = xlds[pl][o];
  }
}

// ---------------------------------------------------------------------------
extern "C" void kernel_launch(void* const* d_in, const int* in_sizes, int n_in,
                              void* d_out, int out_size, void* d_ws, size_t ws_size,
                              hipStream_t stream) {
  (void)in_sizes; (void)n_in; (void)out_size; (void)ws_size;
  const float* pcd       = (const float*)d_in[0];
  const float* neighbors = (const float*)d_in[1];
  const float* q_w       = (const float*)d_in[2];
  const float* k_w       = (const float*)d_in[3];
  const float* v_w       = (const float*)d_in[4];
  float* out = (float*)d_out;
  float* qws = (float*)d_ws;  // 4*8192*64 fp32 = 8 MB scratch for q

  // 1) q projection: 2048 waves (16 points each), 8 waves/block.
  qproj_kernel<<<256, 256, 0, stream>>>(pcd, q_w, qws);
  // 2) attention: one wave per point, 32768 points, 8 waves/block.
  attn_kernel<<<4096, 256, 0, stream>>>(neighbors, k_w, v_w, qws, out);
}
